// GNP_88622355186327
// MI455X (gfx1250) — matrix operand order repaired and checked by reading, not compiled, not run on hardware
//
#include <hip/hip_runtime.h>
#include <math.h>

typedef __attribute__((ext_vector_type(16))) _Float16 v16h;
typedef __attribute__((ext_vector_type(8)))  float    v8f;

#define EMBED_DIM  200
#define NUM_WALKS  25
#define WTRIPLE    (NUM_WALKS * 3)   // 75 walk rows per element per side
#define TILE       16                // batch elements per block
#define THREADS    224               // 7 waves of 32
#define DPAD       208               // padded LDS row stride (dwords)

#if __has_builtin(__builtin_amdgcn_global_load_async_to_lds_b32)
#define HAS_ASYNC_LDS 1
#else
#define HAS_ASYNC_LDS 0
#endif

typedef __attribute__((address_space(1))) int g_int;
typedef __attribute__((address_space(3))) int l_int;

__global__ __launch_bounds__(THREADS)
void gwarmer_score_kernel(const float* __restrict__ emb,
                          const float* __restrict__ layer_w,
                          const int*   __restrict__ uIdx,
                          const int*   __restrict__ iIdx,
                          const int*   __restrict__ uWalks,
                          const int*   __restrict__ iWalks,
                          float*       __restrict__ out,
                          int batch)
{
    __shared__ float sU[TILE][DPAD];      // user reprs (16 x 200, padded)
    __shared__ float sI[TILE][DPAD];      // item reprs
    __shared__ int   sWU[TILE * WTRIPLE]; // staged user walk indices (layers 1..3)
    __shared__ int   sWI[TILE * WTRIPLE]; // staged item walk indices
    __shared__ int   sBU[TILE];           // base user indices
    __shared__ int   sBI[TILE];           // base item indices

    const int tid = threadIdx.x;
    const int b0  = blockIdx.x * TILE;

    // ---- softmax(layer_w) -> coefficients (uniform; scalar loads) ----
    const float w0 = layer_w[0], w1 = layer_w[1], w2 = layer_w[2], w3 = layer_w[3];
    const float mx = fmaxf(fmaxf(w0, w1), fmaxf(w2, w3));
    const float e0 = expf(w0 - mx), e1 = expf(w1 - mx);
    const float e2 = expf(w2 - mx), e3 = expf(w3 - mx);
    const float inv = 1.0f / (e0 + e1 + e2 + e3);
    const float cb  = e0 * inv;                       // base row weight
    const float cL0 = e1 * inv * (1.0f / NUM_WALKS);  // layer-1 per-walk weight
    const float cL1 = e2 * inv * (1.0f / NUM_WALKS);  // layer-2
    const float cL2 = e3 * inv * (1.0f / NUM_WALKS);  // layer-3

    // ---- stage walk indices (layers 1..3) + base indices into LDS ----
    // gfx1250 async global->LDS movers (ASYNCcnt path): no VGPR round-trip,
    // copies overlap with the softmax math above.
    for (int i = tid; i < TILE * WTRIPLE; i += THREADS) {
        const int e = i / WTRIPLE;
        const int r = i % WTRIPLE;
        const int w = r / 3;
        const int l = (r % 3) + 1;                 // walks[:, :, 1:]
        int bb = b0 + e; if (bb >= batch) bb = batch - 1;
        const int g = bb * (NUM_WALKS * 4) + w * 4 + l;
#if HAS_ASYNC_LDS
        __builtin_amdgcn_global_load_async_to_lds_b32(
            (g_int*)(uWalks + g), (l_int*)&sWU[i], 0, 0);
        __builtin_amdgcn_global_load_async_to_lds_b32(
            (g_int*)(iWalks + g), (l_int*)&sWI[i], 0, 0);
#else
        sWU[i] = uWalks[g];
        sWI[i] = iWalks[g];
#endif
    }
    if (tid < TILE) {
        int bb = b0 + tid; if (bb >= batch) bb = batch - 1;
        sBU[tid] = uIdx[bb];
        sBI[tid] = iIdx[bb];
    }
#if HAS_ASYNC_LDS
#if __has_builtin(__builtin_amdgcn_s_wait_asynccnt)
    __builtin_amdgcn_s_wait_asynccnt(0);
#else
    asm volatile("s_wait_asynccnt 0x0" ::: "memory");
#endif
#endif
    __syncthreads();

    // ---- gather + weighted accumulation: thread d handles dim d ----
    const int d = tid;
    if (d < EMBED_DIM) {
        for (int e = 0; e < TILE; ++e) {
            float aU = cb * emb[(unsigned)sBU[e] * EMBED_DIM + d];
            float aI = cb * emb[(unsigned)sBI[e] * EMBED_DIM + d];
            const int* __restrict__ wu = &sWU[e * WTRIPLE];
            const int* __restrict__ wi = &sWI[e * WTRIPLE];
            for (int w = 0; w < NUM_WALKS; ++w) {
                const int j = w * 3;
                if (w + 1 < NUM_WALKS) {
                    // random rows: hardware cannot prefetch; hint next step
                    __builtin_prefetch(&emb[(unsigned)wu[j + 3] * EMBED_DIM + d], 0, 1);
                    __builtin_prefetch(&emb[(unsigned)wi[j + 3] * EMBED_DIM + d], 0, 1);
                }
                aU = fmaf(cL0, emb[(unsigned)wu[j + 0] * EMBED_DIM + d], aU);
                aU = fmaf(cL1, emb[(unsigned)wu[j + 1] * EMBED_DIM + d], aU);
                aU = fmaf(cL2, emb[(unsigned)wu[j + 2] * EMBED_DIM + d], aU);
                aI = fmaf(cL0, emb[(unsigned)wi[j + 0] * EMBED_DIM + d], aI);
                aI = fmaf(cL1, emb[(unsigned)wi[j + 1] * EMBED_DIM + d], aI);
                aI = fmaf(cL2, emb[(unsigned)wi[j + 2] * EMBED_DIM + d], aI);
            }
            sU[e][d] = aU;
            sI[e][d] = aI;
        }
    }
    __syncthreads();

    // ---- wave 0: 16x16 Gram tile via v_wmma_f32_16x16x32_f16, keep diagonal ----
    if (tid < 32) {
        const int  lane = tid;
        const int  mrow = lane & 15;
        const bool hi   = lane >= 16;
        v8f acc = {};
        for (int c = 0; c < 7; ++c) {       // 7 * 32 = 224 >= 200 (zero-padded)
            const int kbase = c * 32;
            v16h a, b;
            #pragma unroll
            for (int h = 0; h < 16; ++h) {
                // A (16x32 f16) layout: VGPR v=h/2; lanes0-15: K in {2v,2v+1} (+16 for v>=4);
                // lanes16-31: +8 within each half-group.
                const int v  = h >> 1;
                const int kA = kbase + ((v < 4) ? 0 : 16) + 2 * (v & 3) + (h & 1) + (hi ? 8 : 0);
                a[h] = (kA < EMBED_DIM) ? (_Float16)sU[mrow][kA] : (_Float16)0.0f;
                // B (32x16 f16) layout: same K order as A (lanes0-15: K=0..7,16..23; lanes16-31: K=8..15,24..31).
                const int kB = kbase + ((h < 8) ? 0 : 8) + h + (hi ? 8 : 0);
                b[h] = (kB < EMBED_DIM) ? (_Float16)sI[mrow][kB] : (_Float16)0.0f;
            }
            acc = __builtin_amdgcn_wmma_f32_16x16x32_f16(
                      false, a, false, b, (short)0, acc, false, false);
        }
        // D layout: VGPR r holds M=r (lanes0-15, N=lane) / M=r+8 (lanes16-31, N=lane-16).
        // Diagonal M==N: lanes 0..7 -> acc[lane]; lanes 24..31 -> acc[lane-24].
        const int nn = lane & 15;
        const int r  = nn - (hi ? 8 : 0);
        if (r >= 0 && r < 8 && (b0 + nn) < batch) {
            float s = 0.0f;
            #pragma unroll
            for (int q = 0; q < 8; ++q)
                if (q == r) s = acc[q];
            out[b0 + nn] = s;
        }
    }
}

extern "C" void kernel_launch(void* const* d_in, const int* in_sizes, int n_in,
                              void* d_out, int out_size, void* d_ws, size_t ws_size,
                              hipStream_t stream) {
    const float* emb = (const float*)d_in[0];
    const float* lw  = (const float*)d_in[1];
    const int*   ui  = (const int*)d_in[2];
    const int*   ii  = (const int*)d_in[3];
    const int*   uw  = (const int*)d_in[4];
    const int*   iw  = (const int*)d_in[5];
    float*       out = (float*)d_out;

    const int batch = in_sizes[2];                 // 8192
    const int grid  = (batch + TILE - 1) / TILE;   // 512 blocks
    gwarmer_score_kernel<<<grid, THREADS, 0, stream>>>(emb, lw, ui, ii, uw, iw, out, batch);
}
